// Net_6253472382984
// MI455X (gfx1250) — matrix operand order repaired and checked by reading, not compiled, not run on hardware
//
#include <hip/hip_runtime.h>
#include <hip/hip_bf16.h>
#include <math.h>

typedef _Float16 f16;
typedef __attribute__((ext_vector_type(8)))  _Float16 v8h;
typedef __attribute__((ext_vector_type(16))) _Float16 v16h;
typedef __attribute__((ext_vector_type(8)))  float    v8f;

#define N_NODES     30000
#define N_EDGES     240000
#define N_FEATS     32
#define E_FEATS     16
#define H_FEAT      64
#define NUM_CLASSES 8
#define NUM_ITER    13
#define N_TILES     1875        // 30000 / 16, exact

// ---------------------------------------------------------------------------
// CDNA5 wave32 WMMA 16x16x32 f16 fragment helpers (ISA 7.12.2 layouts).
// A fragment (16x32, MxK):
//   lanes 0-15 : row = lane,    VGPR0-3 = K0..7,  VGPR4-7 = K16..23
//   lanes16-31 : row = lane-16, VGPR0-3 = K8..15, VGPR4-7 = K24..31
//   -> per lane: two contiguous 8-half (16 B) groups at k0+(half*8) and +16.
// B fragment (32x16, KxN):
//   lanes 0-15 : col = lane,    VGPR0-3 = K0..7,  VGPR4-7 = K16..23
//   lanes16-31 : col = lane-16, VGPR0-3 = K8..15, VGPR4-7 = K24..31
//   -> staged in LDS pre-swizzled so each lane reads 16 contiguous halfs.
// C/D fragment (16x16 f32): VGPR v = row (v + 8*(lane>>4)), col (lane&15)
// ---------------------------------------------------------------------------

__device__ __forceinline__ v16h cat8(v8h lo, v8h hi) {
  return __builtin_shufflevector(lo, hi, 0, 1, 2, 3, 4, 5, 6, 7,
                                         8, 9, 10, 11, 12, 13, 14, 15);
}

// A fragment from a row-major f16 tile in LDS (rows 16B-aligned).
__device__ __forceinline__ v16h frag_a_lds(const f16* base, int ld) {
  const int lane = threadIdx.x & 31;
  const f16* row = base + (lane & 15) * ld + (lane >> 4) * 8;
  return cat8(*(const v8h*)row, *(const v8h*)(row + 16));
}

// A fragment where each lane supplies its own (gathered) row pointer,
// already offset to the K-window start (16B-aligned).
__device__ __forceinline__ v16h frag_a_ptr(const f16* rowp) {
  const f16* p = rowp + ((threadIdx.x & 31) >> 4) * 8;
  return cat8(*(const v8h*)p, *(const v8h*)(p + 16));
}

// Same, converting from an f32 source row (segment-sum accumulator).
__device__ __forceinline__ v16h frag_a_f32ptr(const float* rowp) {
  const float* p = rowp + ((threadIdx.x & 31) >> 4) * 8;
  v16h a;
#pragma unroll
  for (int j = 0; j < 8; ++j) {
    a[j]     = (f16)p[j];
    a[8 + j] = (f16)p[16 + j];
  }
  return a;
}

// B fragment: one aligned 32-byte load from fragment-order swizzled LDS.
__device__ __forceinline__ v16h frag_b_sw(const f16* swz, int frag) {
  const f16* p = swz + ((size_t)frag * 32 + (threadIdx.x & 31)) * 16;
  return *(const v16h*)p;
}

// Stage weight W (row-major [Ksrc x Nsrc] f32) into B-fragment order f16 LDS,
// zero-padding K to STEPS*32 and N to NT*16.
// Layout: swz[((nt*STEPS + s)*32 + lane)*16 + j], k = s*32 + (lane>>4)*8 + (j&7) + (j>>3)*16,
//         n = nt*16 + (lane&15).
template <int NT, int STEPS>
__device__ __forceinline__ void stage_b_swz(f16* swz, const float* __restrict__ W,
                                            int Ksrc, int Nsrc, int tid, int nthr) {
  const int total = NT * STEPS * 32 * 16;
  for (int i = tid; i < total; i += nthr) {
    const int j = i & 15;
    const int lane = (i >> 4) & 31;
    const int f = i >> 9;
    const int s = f % STEPS, nt = f / STEPS;
    const int k = s * 32 + (lane >> 4) * 8 + (j & 7) + (j >> 3) * 16;
    const int n = nt * 16 + (lane & 15);
    swz[i] = (k < Ksrc && n < Nsrc) ? (f16)W[k * Nsrc + n] : (f16)0.0f;
  }
}

__device__ __forceinline__ v8f wmma16(v16h a, v16h b, v8f c) {
  return __builtin_amdgcn_wmma_f32_16x16x32_f16(
      /*neg_a=*/false, a, /*neg_b=*/false, b,
      /*c_mod=*/(short)0, c, /*reuse_a=*/false, /*reuse_b=*/false);
}

__device__ __forceinline__ float sigmoidf_(float x) {
  return 1.0f / (1.0f + __expf(-x));
}

// ---------------------------------------------------------------------------
// Utility kernels
// ---------------------------------------------------------------------------

__global__ void cvt_f32_to_f16(const float* __restrict__ in, f16* __restrict__ out, int n) {
  int i = blockIdx.x * blockDim.x + threadIdx.x;
  if (i < n) out[i] = (f16)in[i];
}

__global__ void zero_f32(float* __restrict__ p, int n) {
  int i = blockIdx.x * blockDim.x + threadIdx.x;
  if (i < n) p[i] = 0.0f;
}

// ---------------------------------------------------------------------------
// GEMM kernels — one 16x16 WMMA tile per wave, exact tiling, LDS epilogues
// ---------------------------------------------------------------------------

// h_n = n_in @ Wn + bn : [30000 x 32] @ [32 x 64] -> f16 [30000 x 64]
// 128 threads = 4 waves (one n-tile each), 16 rows per block, grid = 1875.
__global__ void __launch_bounds__(128) node_fc_kernel(
    const f16* __restrict__ X, const float* __restrict__ W,
    const float* __restrict__ b, f16* __restrict__ H) {
  __shared__ __align__(32) f16 sW[4 * 1 * 32 * 16];
  __shared__ __align__(32) f16 sA[16 * 32];
  __shared__ __align__(32) f16 sO[16 * 64];
  __shared__ float sB[64];
  const int tid = threadIdx.x;
  const int m0 = blockIdx.x * 16;
  stage_b_swz<4, 1>(sW, W, 32, 64, tid, 128);
  if (tid < 64) sB[tid] = b[tid];
  // A tile: 16 rows x 32 halfs = 64 x v8h, fully in-range (30000 % 16 == 0)
  if (tid < 64) ((v8h*)sA)[tid] = ((const v8h*)(X + (size_t)m0 * 32))[tid];
  __syncthreads();
  const int wid = tid >> 5, lane = tid & 31;
  v8f c = {};
  c = wmma16(frag_a_lds(sA, 32), frag_b_sw(sW, wid), c);
  const int n = wid * 16 + (lane & 15);
  const int half = lane >> 4;
#pragma unroll
  for (int v = 0; v < 8; ++v) sO[(v + 8 * half) * 64 + n] = (f16)(c[v] + sB[n]);
  __syncthreads();
  // 16x64 halfs = 128 x v8h, one coalesced 16B store per thread
  const int r = tid >> 3, cc = (tid & 7) * 8;
  *(v8h*)(H + (size_t)(m0 + r) * 64 + cc) = *(const v8h*)(sO + r * 64 + cc);
}

// h_e = e_in @ We + be : [240000 x 16] @ [16 x 64] -> f16 [240000 x 64]
// K zero-padded to 32. 256 threads = 8 waves (2 m-tiles x 4 n-tiles), grid 7500.
// FUSED: also scatters h_e rows into agg[dst] (segment_sum) straight from the
// epilogue registers — avoids a 30 MB/iter re-read of h_e by a separate pass.
__global__ void __launch_bounds__(256) edge_fc_kernel(
    const f16* __restrict__ X, const float* __restrict__ W,
    const float* __restrict__ b, f16* __restrict__ H,
    const int* __restrict__ dst, float* __restrict__ agg) {
  __shared__ __align__(32) f16 sW[4 * 1 * 32 * 16];
  __shared__ __align__(32) f16 sA[32 * 32];
  __shared__ __align__(32) f16 sO[32 * 64];
  __shared__ float sB[64];
  const int tid = threadIdx.x;
  const int m0 = blockIdx.x * 32;
  stage_b_swz<4, 1>(sW, W, 16, 64, tid, 256);
  if (tid < 64) sB[tid] = b[tid];
  // A tile: 32 rows; cols 0..15 from X, 16..31 zero. 32x4 v8h chunks.
  if (tid < 128) {
    const int r = tid >> 2, q = tid & 3;
    v8h val = {};
    if (q < 2) val = *(const v8h*)(X + (size_t)(m0 + r) * 16 + q * 8);
    *(v8h*)(sA + r * 32 + q * 8) = val;
  }
  __syncthreads();
  const int wid = tid >> 5, lane = tid & 31;
  const int mt = wid >> 2, nt = wid & 3;
  v8f c = {};
  c = wmma16(frag_a_lds(sA + mt * 16 * 32, 32), frag_b_sw(sW, nt), c);
  const int n = nt * 16 + (lane & 15);
  const int half = lane >> 4;
#pragma unroll
  for (int v = 0; v < 8; ++v)
    sO[(mt * 16 + v + 8 * half) * 64 + n] = (f16)(c[v] + sB[n]);
  __syncthreads();
  // 32x64 halfs = 256 x v8h: coalesced store of h_e + fused scatter-add
  const int r = tid >> 3, cc = (tid & 7) * 8;
  const v8h val = *(const v8h*)(sO + r * 64 + cc);
  *(v8h*)(H + (size_t)(m0 + r) * 64 + cc) = val;
  float* arow = agg + (size_t)dst[m0 + r] * H_FEAT + cc;
#pragma unroll
  for (int j = 0; j < 8; ++j) atomicAdd(arow + j, (float)val[j]);
}

// e_in = sigmoid(concat(h_n[src], h_e, h_n[dst]) @ Wem + bem)
// [E x 192] @ [192 x 16] -> f16 [E x 16]. A rows gathered per-lane, K = 6 steps.
// 256 threads = 8 waves x 16 edges = 128 edges/block, grid 1875 (exact).
__global__ void __launch_bounds__(256) edge_mpn_kernel(
    const f16* __restrict__ hN, const f16* __restrict__ hE,
    const int* __restrict__ src, const int* __restrict__ dst,
    const float* __restrict__ W /*192x16*/, const float* __restrict__ b /*16*/,
    f16* __restrict__ eOut /*E x 16*/) {
  __shared__ __align__(32) f16 sW[1 * 6 * 32 * 16];
  __shared__ __align__(32) f16 sO[8 * 256];
  __shared__ float sB[16];
  const int tid = threadIdx.x;
  stage_b_swz<1, 6>(sW, W, 192, 16, tid, 256);
  if (tid < 16) sB[tid] = b[tid];
  __syncthreads();
  const int wid = tid >> 5, lane = tid & 31;
  const int e0 = blockIdx.x * 128 + wid * 16;
  const int er = e0 + (lane & 15);              // this lane's A-row edge
  const f16* srcRow = hN + (size_t)src[er] * 64;
  const f16* edgRow = hE + (size_t)er * 64;
  const f16* dstRow = hN + (size_t)dst[er] * 64;
  v8f c = {};
#pragma unroll
  for (int s = 0; s < 6; ++s) {
    const f16* arow = (s < 2) ? (srcRow + 32 * s)
                    : (s < 4) ? (edgRow + 32 * (s - 2))
                              : (dstRow + 32 * (s - 4));
    c = wmma16(frag_a_ptr(arow), frag_b_sw(sW, s), c);
  }
  f16* so = sO + wid * 256;                     // per-wave tile: 16x16, contiguous
  const int n = lane & 15, half = lane >> 4;
#pragma unroll
  for (int v = 0; v < 8; ++v)
    so[(v + 8 * half) * 16 + n] = (f16)sigmoidf_(c[v] + sB[n]);
  // same-wave LDS ordering (DS in-order); 256 halfs = 32 x v8h
  *(v8h*)(eOut + (size_t)e0 * 16 + lane * 8) = *(const v8h*)(so + lane * 8);
}

// n_in = sigmoid(concat(agg, h_n) @ Wnm + bnm) : [N x 128] @ [128 x 32] -> f16 [N x 32]
// 256 threads = 8 waves, one 16-node tile each (tiles 0..1874 exact), grid 235.
__global__ void __launch_bounds__(256) node_mpn_kernel(
    const float* __restrict__ agg /*N x 64*/, const f16* __restrict__ hN /*N x 64*/,
    const float* __restrict__ W /*128x32*/, const float* __restrict__ b /*32*/,
    f16* __restrict__ nOut /*N x 32*/) {
  __shared__ __align__(32) f16 sW[2 * 4 * 32 * 16];
  __shared__ __align__(32) f16 sO[8 * 512];
  __shared__ float sB[32];
  const int tid = threadIdx.x;
  stage_b_swz<2, 4>(sW, W, 128, 32, tid, 256);
  if (tid < 32) sB[tid] = b[tid];
  __syncthreads();
  const int wid = tid >> 5, lane = tid & 31;
  const int tile = blockIdx.x * 8 + wid;
  if (tile < N_TILES) {                         // wave-uniform guard
    const int n0 = tile * 16;
    const int nr = n0 + (lane & 15);
    const float* aggRow = agg + (size_t)nr * 64;
    const f16*   hRow   = hN  + (size_t)nr * 64;
    v8f c0 = {}, c1 = {};
#pragma unroll
    for (int s = 0; s < 4; ++s) {
      v16h a = (s < 2) ? frag_a_f32ptr(aggRow + 32 * s)
                       : frag_a_ptr(hRow + 32 * (s - 2));
      c0 = wmma16(a, frag_b_sw(sW, 0 * 4 + s), c0);
      c1 = wmma16(a, frag_b_sw(sW, 1 * 4 + s), c1);
    }
    f16* so = sO + wid * 512;                   // per-wave tile: 16x32, contiguous
    const int col = lane & 15, half = lane >> 4;
#pragma unroll
    for (int v = 0; v < 8; ++v) {
      const int rl = v + 8 * half;
      so[rl * 32 + col]      = (f16)sigmoidf_(c0[v] + sB[col]);
      so[rl * 32 + col + 16] = (f16)sigmoidf_(c1[v] + sB[col + 16]);
    }
    // 512 halfs = 64 x v8h, 2 per lane
#pragma unroll
    for (int q = 0; q < 2; ++q)
      *(v8h*)(nOut + (size_t)n0 * 32 + q * 256 + lane * 8) =
          *(const v8h*)(so + q * 256 + lane * 8);
  }
}

// out = h_n @ Wfc + bfc : [N x 64] @ [64 x 8] -> f32 [N x 8] (N padded to 16)
__global__ void __launch_bounds__(256) final_fc_kernel(
    const f16* __restrict__ hN, const float* __restrict__ W /*64x8*/,
    const float* __restrict__ b /*8*/, float* __restrict__ out) {
  __shared__ __align__(32) f16 sW[1 * 2 * 32 * 16];
  __shared__ __align__(16) float sOf[8 * 128];
  __shared__ float sB[16];
  const int tid = threadIdx.x;
  stage_b_swz<1, 2>(sW, W, 64, 8, tid, 256);
  if (tid < 16) sB[tid] = (tid < 8) ? b[tid] : 0.0f;
  __syncthreads();
  const int wid = tid >> 5, lane = tid & 31;
  const int tile = blockIdx.x * 8 + wid;
  if (tile < N_TILES) {                         // wave-uniform guard
    const int n0 = tile * 16;
    const int nr = n0 + (lane & 15);
    const f16* hRow = hN + (size_t)nr * 64;
    v8f c = {};
    c = wmma16(frag_a_ptr(hRow +  0), frag_b_sw(sW, 0), c);
    c = wmma16(frag_a_ptr(hRow + 32), frag_b_sw(sW, 1), c);
    float* so = sOf + wid * 128;                // per-wave tile: 16x8 f32, contiguous
    const int col = lane & 15, half = lane >> 4;
    if (col < 8) {
#pragma unroll
      for (int v = 0; v < 8; ++v) so[(v + 8 * half) * 8 + col] = c[v] + sB[col];
    }
    // 128 f32 = 32 x float4
    *(float4*)(out + (size_t)n0 * 8 + lane * 4) = *(const float4*)(so + lane * 4);
  }
}

// ---------------------------------------------------------------------------
// Host launcher
// ---------------------------------------------------------------------------

extern "C" void kernel_launch(void* const* d_in, const int* in_sizes, int n_in,
                              void* d_out, int out_size, void* d_ws, size_t ws_size,
                              hipStream_t stream) {
  (void)in_sizes; (void)n_in; (void)out_size; (void)ws_size;
  const float* node_feat = (const float*)d_in[0];
  const float* edge_feat = (const float*)d_in[1];
  const int*   src       = (const int*)d_in[2];
  const int*   dst       = (const int*)d_in[3];
  const float* Wn  = (const float*)d_in[4];
  const float* bn  = (const float*)d_in[5];
  const float* We  = (const float*)d_in[6];
  const float* be  = (const float*)d_in[7];
  const float* Wem = (const float*)d_in[8];
  const float* bem = (const float*)d_in[9];
  const float* Wnm = (const float*)d_in[10];
  const float* bnm = (const float*)d_in[11];
  const float* Wfc = (const float*)d_in[12];
  const float* bfc = (const float*)d_in[13];
  float* out = (float*)d_out;

  // Scratch layout (~52 MB), 256B-aligned carve-out from d_ws.
  char* ws = (char*)d_ws;
  auto carve = [&](size_t bytes) -> char* {
    char* p = ws;
    ws += (bytes + (size_t)255) & ~(size_t)255;
    return p;
  };
  f16*   nIn = (f16*)carve(sizeof(f16) * (size_t)N_NODES * N_FEATS);
  f16*   eIn = (f16*)carve(sizeof(f16) * (size_t)N_EDGES * E_FEATS);
  f16*   hN  = (f16*)carve(sizeof(f16) * (size_t)N_NODES * H_FEAT);
  f16*   hE  = (f16*)carve(sizeof(f16) * (size_t)N_EDGES * H_FEAT);
  float* agg = (float*)carve(sizeof(float) * (size_t)N_NODES * H_FEAT);

  cvt_f32_to_f16<<<(N_NODES * N_FEATS + 255) / 256, 256, 0, stream>>>(
      node_feat, nIn, N_NODES * N_FEATS);
  cvt_f32_to_f16<<<(N_EDGES * E_FEATS + 255) / 256, 256, 0, stream>>>(
      edge_feat, eIn, N_EDGES * E_FEATS);

  for (int it = 0; it < NUM_ITER; ++it) {
    node_fc_kernel<<<N_TILES, 128, 0, stream>>>(nIn, Wn, bn, hN);
    zero_f32<<<(N_NODES * H_FEAT + 255) / 256, 256, 0, stream>>>(agg, N_NODES * H_FEAT);
    edge_fc_kernel<<<N_EDGES / 32, 256, 0, stream>>>(eIn, We, be, hE, dst, agg);
    edge_mpn_kernel<<<N_EDGES / 128, 256, 0, stream>>>(hN, hE, src, dst, Wem, bem, eIn);
    node_mpn_kernel<<<(N_TILES + 7) / 8, 256, 0, stream>>>(agg, hN, Wnm, bnm, nIn);
  }
  final_fc_kernel<<<(N_TILES + 7) / 8, 256, 0, stream>>>(hN, Wfc, bfc, out);
}
